// GRU_GNN_Model_18253611008144
// MI455X (gfx1250) — matrix operand order repaired and checked by reading, not compiled, not run on hardware
//
#include <hip/hip_runtime.h>
#include <stdint.h>

// ---------------- problem constants (match reference) ----------------
#define BB   16
#define NN   4096
#define EE   65536
#define INF_ 128
#define HH   512
#define GOUT 8
#define F1C  64          // F1 == F2 == 64
#define D1C  512
#define D2C  32768
#define EPSC 1e-5f

typedef __attribute__((ext_vector_type(16))) _Float16 v16h;
typedef __attribute__((ext_vector_type(8)))  float    v8f;

// ---------------- CDNA5 async copy helpers (inline asm, gfx1250) ----------------
// Copy `bytes` (multiple of 512) from contiguous global `src` into LDS at
// byte-offset base `lds_base`, using GLOBAL_LOAD_ASYNC_TO_LDS_B128 (ASYNCcnt).
// One wave32; each lane moves 16B per issue -> 512B per issue.
__device__ __forceinline__ void async_copy_to_lds(unsigned lds_base,
                                                  const void* src, int bytes,
                                                  int lane) {
    for (int off = lane * 16; off < bytes; off += 32 * 16) {
        unsigned           l = lds_base + (unsigned)off;
        unsigned long long g = (unsigned long long)(uintptr_t)((const char*)src + off);
        asm volatile("global_load_async_to_lds_b128 %0, %1, off"
                     :: "v"(l), "v"(g) : "memory");
    }
}

__device__ __forceinline__ void wait_async() {
    asm volatile("s_wait_asynccnt 0x0" ::: "memory");
}

// ---------------- WMMA fragment helpers (wave32, 16x16x32 f16) ----------------
// A: 16x32 (MxK), row-major f32 source (global or LDS), converted to f16.
// lanes 0-15: row M=lane, K = k0+0..15; lanes 16-31: same rows, K = k0+16..31.
__device__ __forceinline__ v16h load_a_frag(const float* A, int lda,
                                            int k0, int lane) {
    int row  = lane & 15;
    int koff = (lane < 16) ? 0 : 16;
    const float* p = A + row * lda + k0 + koff;
    v16h a;
#pragma unroll
    for (int i = 0; i < 8; ++i) a[i]     = (_Float16)p[i];
#pragma unroll
    for (int i = 0; i < 8; ++i) a[8 + i] = (_Float16)p[8 + i];
    return a;
}

// B: 32x16 (KxN) from a [K,N] row-major f32 matrix (stride ldb).
// lanes 0-15: col N=lane, K=k0..k0+15; lanes 16-31: col N=lane-16, K=k0+16..k0+31.
__device__ __forceinline__ v16h load_b_frag_kn(const float* __restrict__ Bm, int ldb,
                                               int k0, int n0, int lane) {
    int col = lane & 15;
    int kb  = (lane < 16) ? 0 : 16;
    const float* p = Bm + (size_t)(k0 + kb) * ldb + n0 + col;
    v16h b;
#pragma unroll
    for (int i = 0; i < 16; ++i) b[i] = (_Float16)p[(size_t)i * ldb];
    return b;
}

// B: 32x16 (KxN) from a [N,K] row-major f32 matrix (torch Linear weight; ldb=K).
__device__ __forceinline__ v16h load_b_frag_nk(const float* __restrict__ Bm, int ldb,
                                               int k0, int n0, int lane) {
    int col = lane & 15;
    int kb  = (lane < 16) ? 0 : 16;
    const float* p = Bm + (size_t)(n0 + col) * ldb + k0 + kb;
    v16h b;
#pragma unroll
    for (int i = 0; i < 16; ++i) b[i] = (_Float16)p[i];
    return b;
}

__device__ __forceinline__ v8f wmma32(v16h a, v16h b, v8f c) {
    // D = A*B + C  (v_wmma_f32_16x16x32_f16, fp32 accumulate)
    return __builtin_amdgcn_wmma_f32_16x16x32_f16(false, a, false, b,
                                                  (short)0, c, false, false);
}

// ---------------- small utility kernels ----------------
__global__ void zero_f(float* __restrict__ p, int cnt) {
    int i = blockIdx.x * blockDim.x + threadIdx.x;
    if (i < cnt) p[i] = 0.0f;
}

__global__ void init_deg(float* __restrict__ deg) {
    int n = blockIdx.x * blockDim.x + threadIdx.x;
    if (n < NN) deg[n] = 1.0f;               // self-loop contribution
}

__global__ void deg_accum(const int* __restrict__ ei, float* __restrict__ deg) {
    int e = blockIdx.x * blockDim.x + threadIdx.x;
    if (e < EE) atomicAdd(&deg[ei[EE + e]], 1.0f);   // dst row of edge_index
}

__global__ void deg_to_dis(float* __restrict__ deg) {
    int n = blockIdx.x * blockDim.x + threadIdx.x;
    if (n < NN) deg[n] = rsqrtf(deg[n]);     // deg >= 1 always (self loop)
}

// hw0[b,n,g] = sum_c pos[b,n,c] * W0[c,g]   (K=3: scalar VALU)
__global__ void gcn0_mm(const float* __restrict__ svp, const float* __restrict__ W0,
                        float* __restrict__ hw0) {
    int tid = blockIdx.x * blockDim.x + threadIdx.x;
    if (tid >= BB * NN * F1C) return;
    int g  = tid & 63;
    int bn = tid >> 6;
    int n  = bn & (NN - 1);
    int b  = bn >> 12;                       // NN = 4096 = 2^12
    const float* p = svp + (size_t)b * (NN * 3) + n * 3;
    hw0[tid] = p[0] * W0[g] + p[1] * W0[F1C + g] + p[2] * W0[2 * F1C + g];
}

// per-edge scatter: agg[b,dst,g] += hw[b,src,g] * dis[src]*dis[dst]
__global__ void edge_agg(const int* __restrict__ ei, const float* __restrict__ dis,
                         const float* __restrict__ hw, float* __restrict__ agg) {
    int e = blockIdx.x;
    int s = ei[e];
    int d = ei[EE + e];
    float w = dis[s] * dis[d];
    const float* src = hw  + ((size_t)s) * F1C;
    float*       dst = agg + ((size_t)d) * F1C;
    for (int idx = threadIdx.x; idx < BB * F1C; idx += blockDim.x) {
        int b = idx >> 6;
        int g = idx & 63;
        atomicAdd(&dst[(size_t)b * NN * F1C + g], src[(size_t)b * NN * F1C + g] * w);
    }
}

// h[b,n,g] = agg[b,n,g] + hw[b,n,g]*dis[n]^2 + bias[g]   (adds self-loop + bias)
__global__ void gcn_finalize(const float* __restrict__ agg, const float* __restrict__ hw,
                             const float* __restrict__ dis, const float* __restrict__ bias,
                             float* __restrict__ out) {
    int tid = blockIdx.x * blockDim.x + threadIdx.x;
    if (tid >= BB * NN * F1C) return;
    int g = tid & 63;
    int n = (tid >> 6) & (NN - 1);
    float di = dis[n];
    out[tid] = agg[tid] + hw[tid] * di * di + bias[g];
}

// ---- WMMA GEMM: (B*N x 64) @ (64 x 64); 1 wave per 16 rows, A staged via async-LDS ----
__global__ __launch_bounds__(32) void gcn_gemm(const float* __restrict__ Hin,
                                               const float* __restrict__ W,
                                               float* __restrict__ Hout) {
    __shared__ float smA[16 * F1C];          // 4 KB tile
    int lane = threadIdx.x;
    const float* A = Hin + (size_t)blockIdx.x * 16 * F1C;
    unsigned lds_base = (unsigned)(uintptr_t)(void*)smA;
    async_copy_to_lds(lds_base, A, 16 * F1C * 4, lane);
    wait_async();

    v8f acc[4] = {{}, {}, {}, {}};
#pragma unroll
    for (int k0 = 0; k0 < F1C; k0 += 32) {
        v16h a = load_a_frag(smA, F1C, k0, lane);
#pragma unroll
        for (int t = 0; t < 4; ++t) {
            v16h b = load_b_frag_kn(W, F1C, k0, t * 16, lane);
            acc[t] = wmma32(a, b, acc[t]);
        }
    }
    float* Cp = Hout + (size_t)blockIdx.x * 16 * F1C;
    int col = lane & 15;
#pragma unroll
    for (int t = 0; t < 4; ++t)
#pragma unroll
        for (int r = 0; r < 8; ++r) {
            int m = (lane < 16) ? r : (r + 8);
            Cp[m * F1C + t * 16 + col] = acc[t][r];
        }
}

// ---------------- WMMA GEMM: M=16 batch GEMMs, A async-staged in LDS ----------------
// A is contiguous [16,K] f32 (lda == K).  NT = 16-wide N-tiles per wave (amortizes A).
// BNK = 0: B is [K,N] row-major (mlp_w*);  BNK = 1: B is [N,K] row-major (w_ih/w_hh).
// epilogue = 0: +bias;  epilogue = 1: +bias, PReLU, BatchNorm(eval).
template <int BNK, int NT>
__global__ __launch_bounds__(32) void wmma_gemm16(
    const float* __restrict__ A,
    const float* __restrict__ Bm, int ldb,
    float* __restrict__ C, int ldc, int K,
    const float* __restrict__ bias, int epilogue,
    const float* __restrict__ alpha, const float* __restrict__ gamma,
    const float* __restrict__ beta, const float* __restrict__ mean,
    const float* __restrict__ var) {
    __shared__ float smA[16 * 512];          // up to 32 KB (K <= 512)
    int lane = threadIdx.x;
    int n0   = blockIdx.x * 16 * NT;

    unsigned lds_base = (unsigned)(uintptr_t)(void*)smA;
    async_copy_to_lds(lds_base, A, 16 * K * 4, lane);
    wait_async();

    v8f acc[NT];
#pragma unroll
    for (int t = 0; t < NT; ++t) acc[t] = (v8f){};

    for (int k0 = 0; k0 < K; k0 += 32) {
        if (k0 + 32 < K) {
            const float* nxt = BNK ? (Bm + (size_t)(n0 + (lane & 15)) * ldb + k0 + 32)
                                   : (Bm + (size_t)(k0 + 32) * ldb + n0 + (lane & 15));
            __builtin_prefetch(nxt, 0, 1);   // global_prefetch_b8
        }
        v16h a = load_a_frag(smA, K, k0, lane);
#pragma unroll
        for (int t = 0; t < NT; ++t) {
            v16h b = BNK ? load_b_frag_nk(Bm, ldb, k0, n0 + t * 16, lane)
                         : load_b_frag_kn(Bm, ldb, k0, n0 + t * 16, lane);
            acc[t] = wmma32(a, b, acc[t]);
        }
    }

#pragma unroll
    for (int t = 0; t < NT; ++t) {
        int   col = n0 + t * 16 + (lane & 15);
        float bv  = bias[col];
        float al = 0.f, ga = 0.f, be = 0.f, me = 0.f, iv = 0.f;
        if (epilogue) {
            al = alpha[col]; ga = gamma[col]; be = beta[col]; me = mean[col];
            iv = rsqrtf(var[col] + EPSC);
        }
#pragma unroll
        for (int r = 0; r < 8; ++r) {
            int   m = (lane < 16) ? r : (r + 8);
            float v = acc[t][r] + bv;
            if (epilogue) {
                v = (v > 0.0f) ? v : al * v;     // PReLU
                v = ga * (v - me) * iv + be;     // BN eval
            }
            C[(size_t)m * ldc + col] = v;
        }
    }
}

// ---------------- GRU gates (torch order r,z,n) ----------------
__global__ void gru_gates(const float* __restrict__ gi, const float* __restrict__ gh,
                          const float* __restrict__ hidden, float* __restrict__ nh) {
    int tid = blockIdx.x * blockDim.x + threadIdx.x;
    if (tid >= BB * HH) return;
    int b = tid / HH, j = tid % HH;
    const float* gib = gi + (size_t)b * 3 * HH;
    const float* ghb = gh + (size_t)b * 3 * HH;
    float r = 1.0f / (1.0f + expf(-(gib[j] + ghb[j])));
    float z = 1.0f / (1.0f + expf(-(gib[HH + j] + ghb[HH + j])));
    float n = tanhf(gib[2 * HH + j] + r * ghb[2 * HH + j]);
    nh[tid] = (1.0f - z) * n + z * hidden[tid];
}

// ---------------- output head: concat([gru_out(8), h1(64)]) @ w_out + b_out ----------
__global__ void head(const float* __restrict__ m2, const float* __restrict__ h1,
                     const float* __restrict__ w_out, const float* __restrict__ b_out,
                     float* __restrict__ y) {
    int tid = blockIdx.x * blockDim.x + threadIdx.x;
    if (tid >= BB * NN) return;
    int b = tid / NN, n = tid % NN;
    float a0 = b_out[0], a1 = b_out[1], a2 = b_out[2];
    const float* g = m2 + (size_t)b * D2C + (size_t)n * GOUT;
#pragma unroll
    for (int j = 0; j < GOUT; ++j) {
        float v = g[j];
        a0 += v * w_out[j * 3 + 0]; a1 += v * w_out[j * 3 + 1]; a2 += v * w_out[j * 3 + 2];
    }
    const float* hp = h1 + ((size_t)b * NN + n) * F1C;
#pragma unroll 8
    for (int k = 0; k < F1C; ++k) {
        float v = hp[k];
        int  j = GOUT + k;
        a0 += v * w_out[j * 3 + 0]; a1 += v * w_out[j * 3 + 1]; a2 += v * w_out[j * 3 + 2];
    }
    float* yp = y + (size_t)b * (NN * 3) + (size_t)n * 3;
    yp[0] = a0; yp[1] = a1; yp[2] = a2;
}

// ---------------- host launcher ----------------
extern "C" void kernel_launch(void* const* d_in, const int* in_sizes, int n_in,
                              void* d_out, int out_size, void* d_ws, size_t ws_size,
                              hipStream_t stream) {
    const float* x        = (const float*)d_in[0];
    const float* svp      = (const float*)d_in[1];
    const float* hidden   = (const float*)d_in[2];
    const int*   ei       = (const int*)  d_in[3];
    const float* W_gcn0   = (const float*)d_in[4];
    const float* b_gcn0   = (const float*)d_in[5];
    const float* W_gcn1   = (const float*)d_in[6];
    const float* b_gcn1   = (const float*)d_in[7];
    const float* w_ih     = (const float*)d_in[8];
    const float* b_ih     = (const float*)d_in[9];
    const float* w_hh     = (const float*)d_in[10];
    const float* b_hh     = (const float*)d_in[11];
    const float* mlp_w1   = (const float*)d_in[12];
    const float* mlp_b1   = (const float*)d_in[13];
    const float* alpha1   = (const float*)d_in[14];
    const float* bn1_g    = (const float*)d_in[15];
    const float* bn1_b    = (const float*)d_in[16];
    const float* bn1_m    = (const float*)d_in[17];
    const float* bn1_v    = (const float*)d_in[18];
    const float* mlp_w2   = (const float*)d_in[19];
    const float* mlp_b2   = (const float*)d_in[20];
    const float* alpha2   = (const float*)d_in[21];
    const float* bn2_g    = (const float*)d_in[22];
    const float* bn2_b    = (const float*)d_in[23];
    const float* bn2_m    = (const float*)d_in[24];
    const float* bn2_v    = (const float*)d_in[25];
    const float* w_out    = (const float*)d_in[26];
    const float* b_out    = (const float*)d_in[27];

    float* y  = (float*)d_out;                 // [B, N*3] = 196608 floats
    float* nh = y + (size_t)BB * NN * 3;       // next_hidden [B, H] = 8192 floats

    // workspace layout (floats)
    float* ws   = (float*)d_ws;
    float* dis  = ws;                                   //   4096
    float* gi   = dis  + NN;                            //  24576
    float* gh   = gi   + BB * 3 * HH;                   //  24576
    float* m1   = gh   + BB * 3 * HH;                   //   8192
    float* m2   = m1   + BB * D1C;                      // 524288
    float* bufA = m2   + (size_t)BB * D2C;              // 4194304 (hw0, later hw1)
    float* bufB = bufA + (size_t)BB * NN * F1C;         // 4194304 (agg0/h0, later agg1/h1)

    const int BNF = BB * NN * F1C;                      // 4194304
    dim3 blk(256);

    // ---- GCN normalization ----
    init_deg<<<(NN + 255) / 256, blk, 0, stream>>>(dis);
    deg_accum<<<(EE + 255) / 256, blk, 0, stream>>>(ei, dis);
    deg_to_dis<<<(NN + 255) / 256, blk, 0, stream>>>(dis);

    // ---- GCN layer 0 ----
    gcn0_mm<<<(BNF + 255) / 256, blk, 0, stream>>>(svp, W_gcn0, bufA);       // hw0
    zero_f<<<(BNF + 255) / 256, blk, 0, stream>>>(bufB, BNF);                // agg0 = 0
    edge_agg<<<EE, blk, 0, stream>>>(ei, dis, bufA, bufB);                   // scatter-add
    gcn_finalize<<<(BNF + 255) / 256, blk, 0, stream>>>(bufB, bufA, dis, b_gcn0, bufB); // h0

    // ---- GCN layer 1 (WMMA GEMM 65536x64x64, async-LDS staged A) ----
    gcn_gemm<<<BB * NN / 16, dim3(32), 0, stream>>>(bufB, W_gcn1, bufA);     // hw1
    zero_f<<<(BNF + 255) / 256, blk, 0, stream>>>(bufB, BNF);                // agg1 = 0
    edge_agg<<<EE, blk, 0, stream>>>(ei, dis, bufA, bufB);
    gcn_finalize<<<(BNF + 255) / 256, blk, 0, stream>>>(bufB, bufA, dis, b_gcn1, bufB); // h1

    // ---- GRUCell (WMMA, M=16) ----
    wmma_gemm16<1, 2><<<3 * HH / 32, dim3(32), 0, stream>>>(
        x, w_ih, INF_, gi, 3 * HH, INF_, b_ih, 0,
        nullptr, nullptr, nullptr, nullptr, nullptr);
    wmma_gemm16<1, 2><<<3 * HH / 32, dim3(32), 0, stream>>>(
        hidden, w_hh, HH, gh, 3 * HH, HH, b_hh, 0,
        nullptr, nullptr, nullptr, nullptr, nullptr);
    gru_gates<<<(BB * HH + 255) / 256, blk, 0, stream>>>(gi, gh, hidden, nh);

    // ---- MLP (WMMA, M=16, fused bias+PReLU+BN epilogue) ----
    wmma_gemm16<0, 1><<<D1C / 16, dim3(32), 0, stream>>>(
        nh, mlp_w1, D1C, m1, D1C, HH, mlp_b1, 1,
        alpha1, bn1_g, bn1_b, bn1_m, bn1_v);
    wmma_gemm16<0, 4><<<D2C / 64, dim3(32), 0, stream>>>(
        m1, mlp_w2, D2C, m2, D2C, D1C, mlp_b2, 1,
        alpha2, bn2_g, bn2_b, bn2_m, bn2_v);

    // ---- output head ----
    head<<<(BB * NN + 255) / 256, blk, 0, stream>>>(m2, bufB, w_out, b_out, y);
}
